// VQAE_49959059587395
// MI455X (gfx1250) — matrix operand order repaired and checked by reading, not compiled, not run on hardware
//
#include <hip/hip_runtime.h>
#include <hip/hip_bf16.h>

typedef __attribute__((ext_vector_type(16))) _Float16 v16h;
typedef __attribute__((ext_vector_type(8)))  _Float16 v8h;
typedef __attribute__((ext_vector_type(2)))  _Float16 v2h;
typedef __attribute__((ext_vector_type(8)))  float    v8f;

#define NROWS 65536
#define DIN   512
#define HID   128
#define DZ    32
#define KCB   1024
#define LDA   40      // LDS row stride in halves for 32-wide K tiles (80B = 5x16B, conflict-spread)

// ---- workspace layout (bytes) ----
#define OFF_H    0u          // h1 / h2 buffer: 65536*128*4 = 33554432
#define OFF_ZF   33554432u   // z  f16: 65536*32*2 = 4194304
#define OFF_ZZ   37748736u   // ||z||^2: 65536*4
#define OFF_CB   38010880u   // codebook f16: 1024*32*2
#define OFF_CC   38076416u   // ||c||^2: 1024*4
#define OFF_TOP  38080512u   // topics: 65536*4
#define OFF_SS   38342656u   // scale1,shift1,scale2,shift2 (4*128 f32)
#define OFF_ACC  38344704u   // sum1,sumsq1,sum2,sumsq2 (512 f32) + zloss + sqerr

__device__ __forceinline__ v16h frag_a(const _Float16* rowp, int g) {
  // 16-bit A 16x32: lane m holds K = g*8 + [0..7]  and  16+g*8 + [0..7]
  v16h a;
  ((v8h*)&a)[0] = *(const v8h*)(rowp + g * 8);
  ((v8h*)&a)[1] = *(const v8h*)(rowp + 16 + g * 8);
  return a;
}
__device__ __forceinline__ v16h frag_b(const _Float16* colp, int g) {
  // B 32x16: lane n holds K = g*8 + [0..7]  and  16+g*8 + [0..7], column-contiguous in LDS/global
  v16h b;
  ((v8h*)&b)[0] = *(const v8h*)(colp + g * 8);
  ((v8h*)&b)[1] = *(const v8h*)(colp + 16 + g * 8);
  return b;
}
__device__ __forceinline__ v8f wmma32(v16h a, v16h b, v8f c) {
  return __builtin_amdgcn_wmma_f32_16x16x32_f16(false, a, false, b, (short)0, c,
                                                false, false);
}

// pack 16 f32 (4x float4) -> 16 f16 and store as two ds_store_b128
__device__ __forceinline__ void cvt_store16(_Float16* dst, const float* src) {
  v8h t0, t1;
#pragma unroll
  for (int i = 0; i < 2; ++i) {
    float4 v = *(const float4*)(src + i * 4);
    t0[i * 4 + 0] = (_Float16)v.x; t0[i * 4 + 1] = (_Float16)v.y;
    t0[i * 4 + 2] = (_Float16)v.z; t0[i * 4 + 3] = (_Float16)v.w;
  }
#pragma unroll
  for (int i = 0; i < 2; ++i) {
    float4 v = *(const float4*)(src + 8 + i * 4);
    t1[i * 4 + 0] = (_Float16)v.x; t1[i * 4 + 1] = (_Float16)v.y;
    t1[i * 4 + 2] = (_Float16)v.z; t1[i * 4 + 3] = (_Float16)v.w;
  }
  *(v8h*)dst = t0;
  *(v8h*)(dst + 8) = t1;
}

// transpose-load a [32 x W] f32 weight slab into LDS [n][k] (stride LDA),
// two adjacent K rows packed per dword -> ds_store_b32 (8 per thread).
// Thread map: k2 = (t>>4)*2, nb = (t&15)*8 ; requires W == 128 columns served
// by srcK (row k) and srcK1 (row k+1).
__device__ __forceinline__ void b_pack_store(_Float16* lb, const float* s0,
                                             const float* s1, int nb, int k2) {
#pragma unroll
  for (int i = 0; i < 2; ++i) {
    float4 a0 = *(const float4*)(s0 + i * 4);
    float4 a1 = *(const float4*)(s1 + i * 4);
    float lo[4] = {a0.x, a0.y, a0.z, a0.w};
    float hi[4] = {a1.x, a1.y, a1.z, a1.w};
#pragma unroll
    for (int j = 0; j < 4; ++j) {
      v2h p;
      p[0] = (_Float16)lo[j];
      p[1] = (_Float16)hi[j];
      *(v2h*)(lb + (nb + i * 4 + j) * LDA + k2) = p;
    }
  }
}

// ------- encoder GEMM1: h1 = X @ W1 + b1 (K=512, N=128), fused BN batch stats -------
__global__ __launch_bounds__(256) void k_gemm_enc1(const float* __restrict__ X,
                                                   const float* __restrict__ w1,
                                                   const float* __restrict__ b1,
                                                   float* __restrict__ h1,
                                                   float* __restrict__ sum,
                                                   float* __restrict__ sumsq) {
  __shared__ __align__(16) _Float16 la[128 * LDA];
  __shared__ __align__(16) _Float16 lb[128 * LDA];
  __shared__ float bsum[128], bsumsq[128];
  const int t = threadIdx.x;
  const int m0 = blockIdx.x * 128;
  const int w = t >> 5, lane = t & 31, n = lane & 15, g = lane >> 4;
  if (t < 128) { bsum[t] = 0.f; bsumsq[t] = 0.f; }
  v8f acc[8] = {};
  for (int kt = 0; kt < DIN / 32; ++kt) {
    const int k0 = kt * 32;
    __syncthreads();
    {  // A tile 128x32 f32 -> f16 LDS [m][k]  (vector ds_store_b128)
      int r = t >> 1, cb = (t & 1) * 16;
      const float* src = X + (size_t)(m0 + r) * DIN + k0 + cb;
      cvt_store16(la + r * LDA + cb, src);
      if (kt + 1 < DIN / 32) __builtin_prefetch(src + 32, 0, 3);
    }
    {  // B tile 32x128 -> lb[n][k], packed-pair stores
      int k2 = (t >> 4) * 2, nb = (t & 15) * 8;
      const float* s0 = w1 + (size_t)(k0 + k2) * HID + nb;
      b_pack_store(lb, s0, s0 + HID, nb, k2);
      if (kt + 1 < DIN / 32) __builtin_prefetch(s0 + 32 * HID, 0, 3);
    }
    __syncthreads();
    v16h a = frag_a(la + (w * 16 + n) * LDA, g);
    v16h bf[8];
#pragma unroll
    for (int j = 0; j < 8; ++j) bf[j] = frag_b(lb + (j * 16 + n) * LDA, g);
#pragma unroll
    for (int j = 0; j < 8; ++j) acc[j] = wmma32(a, bf[j], acc[j]);
  }
#pragma unroll
  for (int j = 0; j < 8; ++j) {
    int col = j * 16 + n;
    float bias = b1[col];
    float s = 0.f, s2 = 0.f;
#pragma unroll
    for (int r = 0; r < 8; ++r) {
      float val = acc[j][r] + bias;
      h1[(size_t)(m0 + w * 16 + g * 8 + r) * HID + col] = val;
      s += val;
      s2 += val * val;
    }
    // partner lane-group holds the other 8 rows of the same column
    s += __shfl_xor(s, 16, 32);
    s2 += __shfl_xor(s2, 16, 32);
    if (g == 0) {  // ds_add_f32 block accumulators
      atomicAdd(&bsum[col], s);
      atomicAdd(&bsumsq[col], s2);
    }
  }
  __syncthreads();
  if (t < 128) {
    atomicAdd(&sum[t], bsum[t]);
    atomicAdd(&sumsq[t], bsumsq[t]);
  }
}

__global__ void k_bnfinal(const float* __restrict__ sum, const float* __restrict__ sumsq,
                          const float* __restrict__ gamma, const float* __restrict__ beta,
                          float* __restrict__ scale, float* __restrict__ shift) {
  int c = threadIdx.x;
  float mu = sum[c] * (1.0f / (float)NROWS);
  float var = sumsq[c] * (1.0f / (float)NROWS) - mu * mu;
  float sc = gamma[c] * rsqrtf(var + 1e-5f);
  scale[c] = sc;
  shift[c] = beta[c] - mu * sc;
}

// BN+ReLU a 16-wide strip of 16 values and store packed f16 (two ds_store_b128)
__device__ __forceinline__ void bn_relu_store16(_Float16* dst, const float* src,
                                                const float* ssc, const float* ssf,
                                                int c0) {
  v8h t0, t1;
#pragma unroll
  for (int i = 0; i < 2; ++i) {
    float4 v = *(const float4*)(src + i * 4);
    float vv[4] = {v.x, v.y, v.z, v.w};
#pragma unroll
    for (int jj = 0; jj < 4; ++jj) {
      int c = c0 + i * 4 + jj;
      t0[i * 4 + jj] = (_Float16)fmaxf(vv[jj] * ssc[c] + ssf[c], 0.0f);
    }
  }
#pragma unroll
  for (int i = 0; i < 2; ++i) {
    float4 v = *(const float4*)(src + 8 + i * 4);
    float vv[4] = {v.x, v.y, v.z, v.w};
#pragma unroll
    for (int jj = 0; jj < 4; ++jj) {
      int c = c0 + 8 + i * 4 + jj;
      t1[i * 4 + jj] = (_Float16)fmaxf(vv[jj] * ssc[c] + ssf[c], 0.0f);
    }
  }
  *(v8h*)dst = t0;
  *(v8h*)(dst + 8) = t1;
}

// ------- encoder GEMM2: z = relu(bn(h1)) @ W2 + b2 (K=128, N=32), emit z f16 + ||z||^2 -------
__global__ __launch_bounds__(256) void k_gemm_enc2(
    const float* __restrict__ h1, const float* __restrict__ w2,
    const float* __restrict__ b2, const float* __restrict__ scale,
    const float* __restrict__ shift, _Float16* __restrict__ zf,
    float* __restrict__ zz) {
  __shared__ __align__(16) _Float16 la[128 * LDA];
  __shared__ __align__(16) _Float16 lb[32 * 136];  // [n][k], stride 136 halves
  __shared__ float ssc[128], ssf[128];
  const int t = threadIdx.x;
  const int m0 = blockIdx.x * 128;
  const int w = t >> 5, lane = t & 31, n = lane & 15, g = lane >> 4;
  if (t < 128) { ssc[t] = scale[t]; ssf[t] = shift[t]; }
  {  // whole B [128,32] -> lb[n][k], packed-pair stores (8x ds_store_b32)
    int k2 = (t >> 2) * 2, nb = (t & 3) * 8;
    const float* s0 = w2 + (size_t)k2 * DZ + nb;
    const float* s1 = s0 + DZ;
#pragma unroll
    for (int i = 0; i < 2; ++i) {
      float4 a0 = *(const float4*)(s0 + i * 4);
      float4 a1 = *(const float4*)(s1 + i * 4);
      float lo[4] = {a0.x, a0.y, a0.z, a0.w};
      float hi[4] = {a1.x, a1.y, a1.z, a1.w};
#pragma unroll
      for (int j = 0; j < 4; ++j) {
        v2h p;
        p[0] = (_Float16)lo[j];
        p[1] = (_Float16)hi[j];
        *(v2h*)(lb + (nb + i * 4 + j) * 136 + k2) = p;
      }
    }
  }
  v8f acc[2] = {};
  for (int kt = 0; kt < 4; ++kt) {
    __syncthreads();
    {  // A tile with fused BN + ReLU
      int r = t >> 1, cb = (t & 1) * 16;
      const float* src = h1 + (size_t)(m0 + r) * HID + kt * 32 + cb;
      bn_relu_store16(la + r * LDA + cb, src, ssc, ssf, kt * 32 + cb);
    }
    __syncthreads();
    v16h a = frag_a(la + (w * 16 + n) * LDA, g);
#pragma unroll
    for (int j = 0; j < 2; ++j) {
      v16h b = frag_b(lb + (j * 16 + n) * 136 + kt * 32, g);
      acc[j] = wmma32(a, b, acc[j]);
    }
  }
  float p[8] = {};
#pragma unroll
  for (int j = 0; j < 2; ++j) {
    int col = j * 16 + n;
    float bias = b2[col];
#pragma unroll
    for (int r = 0; r < 8; ++r) {
      int row = m0 + w * 16 + g * 8 + r;
      float val = acc[j][r] + bias;
      zf[(size_t)row * DZ + col] = (_Float16)val;
      p[r] += val * val;
    }
  }
#pragma unroll
  for (int r = 0; r < 8; ++r) {
#pragma unroll
    for (int m = 1; m < 16; m <<= 1) p[r] += __shfl_xor(p[r], m, 32);
    if (n == 0) zz[m0 + w * 16 + g * 8 + r] = p[r];
  }
}

// ---------------- codebook f16 copy + norms ----------------
__global__ __launch_bounds__(256) void k_cbprep(const float* __restrict__ cb,
                                                _Float16* __restrict__ cb16,
                                                float* __restrict__ cc) {
  int cw = blockIdx.x * 256 + threadIdx.x;
  if (cw >= KCB) return;
  float s = 0.f;
#pragma unroll
  for (int i = 0; i < DZ; ++i) {
    float v = cb[(size_t)cw * DZ + i];
    s += v * v;
    cb16[(size_t)cw * DZ + i] = (_Float16)v;
  }
  cc[cw] = s;
}

// ---------------- nearest codeword: WMMA z.c^T + argmin + sum of min dist ----------------
__global__ __launch_bounds__(256) void k_assign(
    const _Float16* __restrict__ zf, const float* __restrict__ zz,
    const _Float16* __restrict__ cb16, const float* __restrict__ cc,
    int* __restrict__ topics, float* __restrict__ zloss) {
  __shared__ __align__(16) _Float16 la[128 * LDA];
  __shared__ float lzz[128];
  const int t = threadIdx.x;
  const int m0 = blockIdx.x * 128;
  const int w = t >> 5, lane = t & 31, n = lane & 15, g = lane >> 4;
  {  // stage z tile (already f16, row = 32 halves = 4x16B, 2 per thread)
    int r = t >> 1, hb = (t & 1) * 16;
    const uint4* src = (const uint4*)(zf + (size_t)(m0 + r) * DZ + hb);
    uint4* dst = (uint4*)(la + r * LDA + hb);
    dst[0] = src[0];
    dst[1] = src[1];
  }
  if (t < 128) lzz[t] = zz[m0 + t];
  __syncthreads();
  v16h a = frag_a(la + (w * 16 + n) * LDA, g);
  float rowzz[8];
#pragma unroll
  for (int r = 0; r < 8; ++r) rowzz[r] = lzz[w * 16 + g * 8 + r];
  float bestd[8]; int besti[8];
#pragma unroll
  for (int r = 0; r < 8; ++r) { bestd[r] = 3.4e38f; besti[r] = 0; }
  v8f zero = {};
#pragma unroll 2
  for (int ct = 0; ct < KCB / 16; ++ct) {
    int cw = ct * 16 + n;
    v16h b = frag_b(cb16 + (size_t)cw * DZ, g);  // L2-hot, column-contiguous
    v8f d = wmma32(a, b, zero);
    float ccn = cc[cw];
#pragma unroll
    for (int r = 0; r < 8; ++r) {
      float dist = rowzz[r] + ccn - 2.0f * d[r];
      if (dist < bestd[r]) { bestd[r] = dist; besti[r] = cw; }
    }
  }
  float lsum = 0.f;
#pragma unroll
  for (int r = 0; r < 8; ++r) {
#pragma unroll
    for (int m = 1; m < 16; m <<= 1) {
      float od = __shfl_xor(bestd[r], m, 32);
      int oi = __shfl_xor(besti[r], m, 32);
      if (od < bestd[r]) { bestd[r] = od; besti[r] = oi; }
    }
    if (n == 0) {
      topics[m0 + w * 16 + g * 8 + r] = besti[r];
      lsum += bestd[r];
    }
  }
  if (n == 0) atomicAdd(zloss, lsum);
}

// ---- decoder GEMM1: h2 = cb[topics] @ W1d + b1d (K=32, N=128), fused BN stats ----
__global__ __launch_bounds__(256) void k_gemm_dec1(
    const int* __restrict__ topics, const _Float16* __restrict__ cb16,
    const float* __restrict__ w1, const float* __restrict__ b1,
    float* __restrict__ h2, float* __restrict__ sum, float* __restrict__ sumsq) {
  __shared__ __align__(16) _Float16 la[128 * LDA];
  __shared__ __align__(16) _Float16 lb[128 * LDA];
  __shared__ int ltop[128];
  __shared__ float bsum[128], bsumsq[128];
  const int t = threadIdx.x;
  const int m0 = blockIdx.x * 128;
  const int w = t >> 5, lane = t & 31, n = lane & 15, g = lane >> 4;
  if (t < 128) {
    ltop[t] = topics[m0 + t];
    bsum[t] = 0.f;
    bsumsq[t] = 0.f;
  }
  {  // B [32,128] -> lb[n][k], packed-pair stores
    int k2 = (t >> 4) * 2, nb = (t & 15) * 8;
    const float* s0 = w1 + (size_t)k2 * HID + nb;
    b_pack_store(lb, s0, s0 + HID, nb, k2);
  }
  __syncthreads();
  {  // gather selected codewords into A tile
    int r = t >> 1, hb = (t & 1) * 16;
    const uint4* src = (const uint4*)(cb16 + (size_t)ltop[r] * DZ + hb);
    uint4* dst = (uint4*)(la + r * LDA + hb);
    dst[0] = src[0];
    dst[1] = src[1];
  }
  __syncthreads();
  v16h a = frag_a(la + (w * 16 + n) * LDA, g);
  v8f zero = {};
  v16h bf[8];
#pragma unroll
  for (int j = 0; j < 8; ++j) bf[j] = frag_b(lb + (j * 16 + n) * LDA, g);
#pragma unroll
  for (int j = 0; j < 8; ++j) {
    v8f d = wmma32(a, bf[j], zero);
    int col = j * 16 + n;
    float bias = b1[col];
    float s = 0.f, s2 = 0.f;
#pragma unroll
    for (int r = 0; r < 8; ++r) {
      float val = d[r] + bias;
      h2[(size_t)(m0 + w * 16 + g * 8 + r) * HID + col] = val;
      s += val;
      s2 += val * val;
    }
    s += __shfl_xor(s, 16, 32);
    s2 += __shfl_xor(s2, 16, 32);
    if (g == 0) {
      atomicAdd(&bsum[col], s);
      atomicAdd(&bsumsq[col], s2);
    }
  }
  __syncthreads();
  if (t < 128) {
    atomicAdd(&sum[t], bsum[t]);
    atomicAdd(&sumsq[t], bsumsq[t]);
  }
}

// ------- decoder GEMM2 + residual: sqerr += sum((relu(bn(h2))@W2d + b2d - X)^2) -------
__global__ __launch_bounds__(256) void k_gemm_dec2(
    const float* __restrict__ h2, const float* __restrict__ w2,
    const float* __restrict__ b2, const float* __restrict__ scale,
    const float* __restrict__ shift, const float* __restrict__ X,
    float* __restrict__ sqerr) {
  __shared__ __align__(16) _Float16 la[128 * LDA];
  __shared__ __align__(16) _Float16 lb[128 * LDA];
  __shared__ float ssc[128], ssf[128];
  __shared__ float red[256];
  const int t = threadIdx.x;
  const int m0 = (blockIdx.x >> 2) * 128;
  const int n0 = (blockIdx.x & 3) * 128;
  const int w = t >> 5, lane = t & 31, n = lane & 15, g = lane >> 4;
  if (t < 128) { ssc[t] = scale[t]; ssf[t] = shift[t]; }
  v8f acc[8] = {};
  for (int kt = 0; kt < 4; ++kt) {
    __syncthreads();
    {  // A with fused BN + ReLU (vector ds_store_b128)
      int r = t >> 1, cb = (t & 1) * 16;
      const float* src = h2 + (size_t)(m0 + r) * HID + kt * 32 + cb;
      bn_relu_store16(la + r * LDA + cb, src, ssc, ssf, kt * 32 + cb);
    }
    {  // B chunk [32 x 128] of W2d, packed-pair stores
      int k2 = (t >> 4) * 2, nb = (t & 15) * 8;
      const float* s0 = w2 + (size_t)(kt * 32 + k2) * DIN + n0 + nb;
      b_pack_store(lb, s0, s0 + DIN, nb, k2);
      if (kt < 3) __builtin_prefetch(s0 + 32 * DIN, 0, 3);
    }
    __syncthreads();
    v16h a = frag_a(la + (w * 16 + n) * LDA, g);
    v16h bf[8];
#pragma unroll
    for (int j = 0; j < 8; ++j) bf[j] = frag_b(lb + (j * 16 + n) * LDA, g);
#pragma unroll
    for (int j = 0; j < 8; ++j) acc[j] = wmma32(a, bf[j], acc[j]);
  }
  float p = 0.f;
#pragma unroll
  for (int j = 0; j < 8; ++j) {
    int col = n0 + j * 16 + n;
    float bias = b2[col];
#pragma unroll
    for (int r = 0; r < 8; ++r) {
      int row = m0 + w * 16 + g * 8 + r;
      float diff = acc[j][r] + bias - X[(size_t)row * DIN + col];
      p += diff * diff;
    }
  }
  red[t] = p;
  __syncthreads();
  for (int s = 128; s > 0; s >>= 1) {
    if (t < s) red[t] += red[t + s];
    __syncthreads();
  }
  if (t == 0) atomicAdd(sqerr, red[0]);
}

__global__ void k_final(const float* __restrict__ zloss,
                        const float* __restrict__ sqerr,
                        float* __restrict__ out) {
  out[0] = 2.0f * zloss[0] + sqrtf(sqerr[0]);
}

extern "C" void kernel_launch(void* const* d_in, const int* in_sizes, int n_in,
                              void* d_out, int out_size, void* d_ws, size_t ws_size,
                              hipStream_t stream) {
  (void)in_sizes; (void)n_in; (void)out_size; (void)ws_size;
  const float* X       = (const float*)d_in[0];
  const float* enc_w1  = (const float*)d_in[1];
  const float* enc_b1  = (const float*)d_in[2];
  const float* enc_g1  = (const float*)d_in[3];
  const float* enc_be1 = (const float*)d_in[4];
  const float* enc_w2  = (const float*)d_in[5];
  const float* enc_b2  = (const float*)d_in[6];
  const float* dec_w1  = (const float*)d_in[7];
  const float* dec_b1  = (const float*)d_in[8];
  const float* dec_g1  = (const float*)d_in[9];
  const float* dec_be1 = (const float*)d_in[10];
  const float* dec_w2  = (const float*)d_in[11];
  const float* dec_b2  = (const float*)d_in[12];
  const float* cbf     = (const float*)d_in[13];

  char* ws = (char*)d_ws;
  float*    h_buf  = (float*)(ws + OFF_H);
  _Float16* zf     = (_Float16*)(ws + OFF_ZF);
  float*    zz     = (float*)(ws + OFF_ZZ);
  _Float16* cb16   = (_Float16*)(ws + OFF_CB);
  float*    cc     = (float*)(ws + OFF_CC);
  int*      topics = (int*)(ws + OFF_TOP);
  float*    scale1 = (float*)(ws + OFF_SS);
  float*    shift1 = scale1 + 128;
  float*    scale2 = scale1 + 256;
  float*    shift2 = scale1 + 384;
  float*    sum1   = (float*)(ws + OFF_ACC);
  float*    sumsq1 = sum1 + 128;
  float*    sum2   = sum1 + 256;
  float*    sumsq2 = sum1 + 384;
  float*    zloss  = sum1 + 512;
  float*    sqerr  = sum1 + 513;

  hipMemsetAsync((void*)sum1, 0, (512 + 2) * sizeof(float), stream);

  k_gemm_enc1<<<NROWS / 128, 256, 0, stream>>>(X, enc_w1, enc_b1, h_buf, sum1, sumsq1);
  k_bnfinal<<<1, 128, 0, stream>>>(sum1, sumsq1, enc_g1, enc_be1, scale1, shift1);
  k_cbprep<<<KCB / 256, 256, 0, stream>>>(cbf, cb16, cc);
  k_gemm_enc2<<<NROWS / 128, 256, 0, stream>>>(h_buf, enc_w2, enc_b2, scale1, shift1, zf, zz);
  k_assign<<<NROWS / 128, 256, 0, stream>>>(zf, zz, cb16, cc, topics, zloss);
  k_gemm_dec1<<<NROWS / 128, 256, 0, stream>>>(topics, cb16, dec_w1, dec_b1, h_buf, sum2, sumsq2);
  k_bnfinal<<<1, 128, 0, stream>>>(sum2, sumsq2, dec_g1, dec_be1, scale2, shift2);
  k_gemm_dec2<<<(NROWS / 128) * 4, 256, 0, stream>>>(h_buf, dec_w2, dec_b2, scale2, shift2, X, sqerr);
  k_final<<<1, 1, 0, stream>>>(zloss, sqerr, (float*)d_out);
}
